// CrissCrossAttention_65816078844146
// MI455X (gfx1250) — compile-verified, matrix operand order checked by reading, not run on hardware
//
#include <hip/hip_runtime.h>

#define B_    8
#define C_    512
#define IC_   64
#define H_    96
#define W_    96
#define HW_   (H_*W_)
#define P_    (B_*HW_)     // 73728 pixel-rows
#define NQKV_ 640          // 64 q + 64 k + 512 v channels
#define LOGD_ 192          // H + W logits per pixel

typedef __bf16 bf16;
typedef float v8f   __attribute__((ext_vector_type(8)));
typedef bf16  v16bf __attribute__((ext_vector_type(16)));
typedef bf16  v8bf  __attribute__((ext_vector_type(8)));

union V16 { v16bf v; v8bf h[2]; };

// D = A(16x32 bf16) * B(32x16 bf16) + C(16x16 f32)
static __device__ __forceinline__ v8f wmma_bf16(v16bf a, v16bf b, v8f c) {
  return __builtin_amdgcn_wmma_f32_16x16x32_bf16(false, a, false, b, (short)0, c,
                                                 false, false);
}

// Per-lane operand fragment: elems [0..7] = K(koff..koff+7), [8..15] = K(koff+16..+23).
// p points at (K = kb + koff); both 16B chunks must be 16B aligned.
static __device__ __forceinline__ v16bf ld_bf16x16(const bf16* p) {
  V16 u;
  u.h[0] = *reinterpret_cast<const v8bf*>(p);
  u.h[1] = *reinterpret_cast<const v8bf*>(p + 16);
  return u.v;
}

// Same fragment but source is fp32 (attention probs), converted to bf16 in regs.
static __device__ __forceinline__ v16bf ld_f32x16_cvt(const float* p) {
  V16 u;
#pragma unroll
  for (int i = 0; i < 8; ++i) { u.v[i] = (bf16)p[i]; u.v[i + 8] = (bf16)p[i + 16]; }
  return u.v;
}

// ---------------- Kernel 1: x (B,C,HW) f32 -> xT (B,HW,C) bf16 ----------------
__global__ void __launch_bounds__(256) k_transpose_cvt(const float* __restrict__ x,
                                                       bf16* __restrict__ xT) {
  __shared__ float tile[32][33];
  int b  = blockIdx.z;
  int p0 = blockIdx.x * 32;
  int c0 = blockIdx.y * 32;
#pragma unroll
  for (int i = 0; i < 4; ++i)
    tile[threadIdx.y + 8*i][threadIdx.x] =
        x[((size_t)b*C_ + c0 + threadIdx.y + 8*i) * HW_ + p0 + threadIdx.x];
  __syncthreads();
#pragma unroll
  for (int i = 0; i < 4; ++i)
    xT[((size_t)b*HW_ + p0 + threadIdx.y + 8*i) * C_ + c0 + threadIdx.x] =
        (bf16)tile[threadIdx.x][threadIdx.y + 8*i];
}

// ---------------- Kernel 2: pack [Wq;Wk;Wv] -> Wc (640,512) bf16 ----------------
__global__ void k_pack_weights(const float* __restrict__ Wq, const float* __restrict__ Wk,
                               const float* __restrict__ Wv, bf16* __restrict__ Wc) {
  int idx = blockIdx.x * blockDim.x + threadIdx.x;
  if (idx >= NQKV_ * C_) return;
  int r = idx / C_, c = idx % C_;
  float v;
  if (r < IC_)        v = Wq[r * C_ + c];
  else if (r < 2*IC_) v = Wk[(r - IC_) * C_ + c];
  else                v = Wv[(r - 2*IC_) * C_ + c];
  Wc[idx] = (bf16)v;
}

// ------------- Kernel 3: fused QKV GEMM: qkv[p,n] = xT[p,:].Wc[n,:] + bias -------------
__global__ void __launch_bounds__(256) k_qkv_gemm(
    const bf16* __restrict__ xT, const bf16* __restrict__ Wc,
    const float* __restrict__ bq, const float* __restrict__ bk,
    const float* __restrict__ bv, bf16* __restrict__ qkv) {
  int wv   = threadIdx.x >> 5;
  int lane = threadIdx.x & 31;
  int ln   = lane & 15;
  int koff = (lane < 16) ? 0 : 8;
  int rsel = (lane >> 4) * 8;
  int row0 = blockIdx.x * 128 + wv * 16;
  int col0 = blockIdx.y * 64;

  v8f acc[4] = {};
  const bf16* arow = xT + (size_t)(row0 + ln) * C_ + koff;
  for (int kb = 0; kb < C_; kb += 32) {
    __builtin_prefetch(arow + kb + 64, 0, 0);
    v16bf a = ld_bf16x16(arow + kb);
#pragma unroll
    for (int t = 0; t < 4; ++t) {
      const bf16* bp = Wc + (size_t)(col0 + t*16 + ln) * C_ + kb + koff;
      acc[t] = wmma_bf16(a, ld_bf16x16(bp), acc[t]);
    }
  }
#pragma unroll
  for (int t = 0; t < 4; ++t) {
    int col = col0 + t*16 + ln;
    float bias = (col < IC_) ? bq[col] : (col < 2*IC_) ? bk[col - IC_] : bv[col - 2*IC_];
#pragma unroll
    for (int r = 0; r < 8; ++r) {
      int row = row0 + r + rsel;
      qkv[(size_t)row * NQKV_ + col] = (bf16)(acc[t][r] + bias);
    }
  }
}

// ------------- Kernel 4: attention scores (mode 0: column/e_h, mode 1: row/e_w) -------------
__global__ void __launch_bounds__(192) k_scores(const bf16* __restrict__ qkv,
                                                float* __restrict__ logits) {
  int line = blockIdx.x;            // w (mode 0) or h (mode 1)
  int b    = blockIdx.y;
  int mode = blockIdx.z;
  int wv   = threadIdx.x >> 5;
  int lane = threadIdx.x & 31;
  int ln   = lane & 15;
  int koff = (lane < 16) ? 0 : 8;
  int rsel = (lane >> 4) * 8;

  int mg = wv * 16 + ln;            // global row index within 96
  size_t pA = (mode == 0) ? ((size_t)mg * W_ + line) : ((size_t)line * W_ + mg);
  const bf16* aptr = qkv + ((size_t)b * HW_ + pA) * NQKV_ + koff;     // q channels 0..63
  v16bf a0 = ld_bf16x16(aptr);
  v16bf a1 = ld_bf16x16(aptr + 32);

#pragma unroll 1
  for (int nt = 0; nt < 6; ++nt) {
    int ng = nt * 16 + ln;
    size_t pB = (mode == 0) ? ((size_t)ng * W_ + line) : ((size_t)line * W_ + ng);
    const bf16* bptr = qkv + ((size_t)b * HW_ + pB) * NQKV_ + IC_ + koff;  // k channels
    v8f acc = {};
    acc = wmma_bf16(a0, ld_bf16x16(bptr), acc);
    acc = wmma_bf16(a1, ld_bf16x16(bptr + 32), acc);
#pragma unroll
    for (int r = 0; r < 8; ++r) {
      int mrow = wv * 16 + r + rsel;
      int ncol = nt * 16 + ln;
      float val = acc[r];
      if (mode == 0) {
        if (ncol == mrow) val = -__builtin_inff();   // mask diagonal of e_h
        logits[(((size_t)b * H_ + mrow) * W_ + line) * LOGD_ + ncol] = val;
      } else {
        logits[(((size_t)b * H_ + line) * W_ + mrow) * LOGD_ + H_ + ncol] = val;
      }
    }
  }
}

// ------------- Kernel 5: softmax over 192 logits per pixel (one wave / pixel) -------------
__global__ void __launch_bounds__(256) k_softmax(float* __restrict__ logits) {
  int wv   = threadIdx.x >> 5;
  int lane = threadIdx.x & 31;
  size_t px = (size_t)blockIdx.x * 8 + wv;
  float* p = logits + px * LOGD_;
  float v[6];
  float mx = -__builtin_inff();
#pragma unroll
  for (int j = 0; j < 6; ++j) { v[j] = p[lane + 32*j]; mx = fmaxf(mx, v[j]); }
#pragma unroll
  for (int s = 16; s > 0; s >>= 1) mx = fmaxf(mx, __shfl_xor(mx, s, 32));
  float sum = 0.f;
#pragma unroll
  for (int j = 0; j < 6; ++j) { v[j] = __expf(v[j] - mx); sum += v[j]; }
#pragma unroll
  for (int s = 16; s > 0; s >>= 1) sum += __shfl_xor(sum, s, 32);
  float inv = 1.f / sum;
#pragma unroll
  for (int j = 0; j < 6; ++j) p[lane + 32*j] = v[j] * inv;
}

// ------------- Kernel 6: out_h[b,h,w,c] = sum_g a_h[h,w,g] * v[g,w,c] (per (b,w)) -------------
__global__ void __launch_bounds__(192) k_agg_col(const bf16* __restrict__ qkv,
                                                 const float* __restrict__ probs,
                                                 bf16* __restrict__ oh) {
  __shared__ bf16 Vt[64 * 104];      // Vt[c_local][g], 16B-aligned rows (pitch 104)
  int w = blockIdx.x, b = blockIdx.y;
  int wv   = threadIdx.x >> 5;
  int lane = threadIdx.x & 31;
  int ln   = lane & 15;
  int koff = (lane < 16) ? 0 : 8;
  int rsel = (lane >> 4) * 8;

  for (int ct = 0; ct < 8; ++ct) {
    int cb = ct * 64;
    __syncthreads();
    for (int i = threadIdx.x; i < 96 * 64; i += 192) {   // stage V column slice, transposed
      int g = i >> 6, cc = i & 63;
      Vt[cc * 104 + g] =
          qkv[((size_t)b * HW_ + (size_t)g * W_ + w) * NQKV_ + 2*IC_ + cb + cc];
    }
    __syncthreads();

    int h_lane = wv * 16 + ln;
    const float* arow = probs + (((size_t)b * H_ + h_lane) * W_ + w) * LOGD_ + koff;
    v16bf a[3];
#pragma unroll
    for (int ks = 0; ks < 3; ++ks) a[ks] = ld_f32x16_cvt(arow + ks * 32);
#pragma unroll 1
    for (int nt = 0; nt < 4; ++nt) {
      v8f acc = {};
#pragma unroll
      for (int ks = 0; ks < 3; ++ks)
        acc = wmma_bf16(a[ks], ld_bf16x16(&Vt[(nt*16 + ln) * 104 + ks*32 + koff]), acc);
#pragma unroll
      for (int r = 0; r < 8; ++r) {
        int hh = wv * 16 + r + rsel;
        oh[(((size_t)b * H_ + hh) * W_ + w) * C_ + cb + nt*16 + ln] = (bf16)acc[r];
      }
    }
  }
}

// ------- Kernel 7: out_w + epilogue: out = gamma*(out_h+out_w) + x  (per (b,h)) -------
__global__ void __launch_bounds__(192) k_agg_row_final(
    const bf16* __restrict__ qkv, const float* __restrict__ probs,
    const bf16* __restrict__ oh, const float* __restrict__ x,
    const float* __restrict__ gamma, float* __restrict__ out) {
  __shared__ bf16  Vt[64 * 104];     // Vt[c_local][v']
  __shared__ float Res[64 * 100];    // Res[c_local][w]
  int h = blockIdx.x, b = blockIdx.y;
  int wv   = threadIdx.x >> 5;
  int lane = threadIdx.x & 31;
  int ln   = lane & 15;
  int koff = (lane < 16) ? 0 : 8;
  int rsel = (lane >> 4) * 8;
  float gm = gamma[0];

  for (int ct = 0; ct < 8; ++ct) {
    int cb = ct * 64;
    __syncthreads();
    for (int i = threadIdx.x; i < 96 * 64; i += 192) {   // stage V row slice, transposed
      int vv = i >> 6, cc = i & 63;
      Vt[cc * 104 + vv] =
          qkv[((size_t)b * HW_ + (size_t)h * W_ + vv) * NQKV_ + 2*IC_ + cb + cc];
    }
    __syncthreads();

    int w_lane = wv * 16 + ln;
    const float* arow =
        probs + (((size_t)b * H_ + h) * W_ + w_lane) * LOGD_ + H_ + koff;  // a_w part
    v16bf a[3];
#pragma unroll
    for (int ks = 0; ks < 3; ++ks) a[ks] = ld_f32x16_cvt(arow + ks * 32);
#pragma unroll 1
    for (int nt = 0; nt < 4; ++nt) {
      v8f acc = {};
#pragma unroll
      for (int ks = 0; ks < 3; ++ks)
        acc = wmma_bf16(a[ks], ld_bf16x16(&Vt[(nt*16 + ln) * 104 + ks*32 + koff]), acc);
#pragma unroll
      for (int r = 0; r < 8; ++r) {
        int ww = wv * 16 + r + rsel;
        int cc = nt * 16 + ln;
        float ohv = (float)oh[(((size_t)b * H_ + h) * W_ + ww) * C_ + cb + cc];
        Res[cc * 100 + ww] = gm * (acc[r] + ohv);
      }
    }
    __syncthreads();
    for (int i = threadIdx.x; i < 64 * 96; i += 192) {   // coalesced (B,C,H,W) stores
      int cc = i / 96, ww = i % 96;
      size_t o = (((size_t)b * C_ + cb + cc) * H_ + h) * W_ + ww;
      out[o] = Res[cc * 100 + ww] + x[o];
    }
  }
}

extern "C" void kernel_launch(void* const* d_in, const int* in_sizes, int n_in,
                              void* d_out, int out_size, void* d_ws, size_t ws_size,
                              hipStream_t stream) {
  const float* x     = (const float*)d_in[0];
  const float* Wq    = (const float*)d_in[1];
  const float* bq    = (const float*)d_in[2];
  const float* Wk    = (const float*)d_in[3];
  const float* bk    = (const float*)d_in[4];
  const float* Wv    = (const float*)d_in[5];
  const float* bv    = (const float*)d_in[6];
  const float* gamma = (const float*)d_in[7];
  float* out = (float*)d_out;

  // Workspace layout (bytes):
  //   [0)            xT  bf16 (B,HW,C)      75,497,472   (reused as out_h after GEMM)
  //   [75,497,472)   qkv bf16 (B,HW,640)    94,371,840
  //   [169,869,312)  logits f32 (B,H,W,192) 56,623,104
  //   [226,492,416)  Wc  bf16 (640,512)        655,360   -> total 227,147,776
  char* ws = (char*)d_ws;
  bf16*  xT     = (bf16*)(ws);
  bf16*  qkv    = (bf16*)(ws + 75497472ull);
  float* logits = (float*)(ws + 169869312ull);
  bf16*  Wc     = (bf16*)(ws + 226492416ull);
  bf16*  oh     = xT;   // alias: xT is dead after the QKV GEMM

  k_transpose_cvt<<<dim3(HW_/32, C_/32, B_), dim3(32, 8), 0, stream>>>(x, xT);
  k_pack_weights<<<(NQKV_*C_ + 255)/256, 256, 0, stream>>>(Wq, Wk, Wv, Wc);
  k_qkv_gemm<<<dim3(P_/128, NQKV_/64), 256, 0, stream>>>(xT, Wc, bq, bk, bv, qkv);
  k_scores<<<dim3(W_, B_, 2), 192, 0, stream>>>(qkv, logits);
  k_softmax<<<P_/8, 256, 0, stream>>>(logits);
  k_agg_col<<<dim3(W_, B_), 192, 0, stream>>>(qkv, logits, oh);
  k_agg_row_final<<<dim3(H_, B_), 192, 0, stream>>>(qkv, logits, oh, x, gamma, out);
}
